// RelationEnhanceAttention_85779086836345
// MI455X (gfx1250) — compile-verified, matrix operand order checked by reading, not run on hardware
//
#include <hip/hip_runtime.h>
#include <hip/hip_bf16.h>
#include <math.h>

// ---------------------------------------------------------------------------
// RelationEnhanceAttention, algebraically restructured for MI455X (gfx1250).
//   B=32, N=49, D=512, H=8, DK=64, NREL=TOPK=14
// All dense GEMMs: f16 operands (K-innermost both sides), fp32 accumulate on
// v_wmma_f32_16x16x32_f16. 16x64 C-tile per wave, pure b128 fragment loads.
// ---------------------------------------------------------------------------

typedef __attribute__((ext_vector_type(16))) _Float16 v16h;
typedef __attribute__((ext_vector_type(8)))  _Float16 v8h;
typedef __attribute__((ext_vector_type(8)))  float    v8f;

#define BQ      32
#define NQ      49
#define DMODEL  512
#define NHEAD   8
#define DKH     64
#define NRELC   14
#define ROWS    (BQ * NQ)            // 1568
#define KCONV   (DMODEL * 9)         // 4608
#define NPAIR   (BQ * NQ * NQ)       // 76832
#define LN_EPS  1e-5f

// ---------------------------------------------------------------------------
// WMMA GEMM: C[M,N] = A[M,K] * B^T[N,K] (+bias[N]).
// A: f16 row-major (M,K), lda mult of 8. Bt: f16 row-major (N,K), ldb mult 8.
// REQUIRES: M%16==0, N%64==0, K%32==0 (true for every call site here).
// One wave32 computes a 16x64 C tile; K stepped by 32.
// Fragment layout (ISA 16-bit A 16x32): halves 0..7  <- K = k0+8*half+0..7
//                                       halves 8..15 <- K = k0+8*half+16..23
// i.e. two contiguous 16-byte segments per lane -> two global_load_b128 each.
// ---------------------------------------------------------------------------
__global__ void gemm_wmma_f16(const _Float16* __restrict__ A,
                              const _Float16* __restrict__ Bt,
                              const float* __restrict__ bias,
                              float* __restrict__ C,
                              int M, int N, int K, int lda, int ldb, int ldc) {
  const int ntn  = N >> 6;                       // 64-wide column tiles
  const int tile = blockIdx.x * blockDim.y + threadIdx.y;
  const int tm   = tile / ntn;
  const int tn   = tile - tm * ntn;
  const int row0 = tm << 4;
  const int col0 = tn << 6;
  if (row0 >= M) return;                         // wave-uniform guard only

  const int lane = threadIdx.x;                  // 0..31
  const int hf   = lane >> 4;                    // 0/1
  const int idx  = lane & 15;

  const _Float16* pa = A + (size_t)(row0 + idx) * lda + hf * 8;
  const _Float16* pb0 = Bt + (size_t)(col0 + 0  + idx) * ldb + hf * 8;
  const _Float16* pb1 = Bt + (size_t)(col0 + 16 + idx) * ldb + hf * 8;
  const _Float16* pb2 = Bt + (size_t)(col0 + 32 + idx) * ldb + hf * 8;
  const _Float16* pb3 = Bt + (size_t)(col0 + 48 + idx) * ldb + hf * 8;

  v8f acc0 = {}, acc1 = {}, acc2 = {}, acc3 = {};
  union F16x16 { v16h v; v8h h[2]; };

  for (int k0 = 0; k0 < K; k0 += 32) {
    F16x16 af, b0, b1, b2, b3;
    af.h[0] = *(const v8h*)(pa + k0);
    af.h[1] = *(const v8h*)(pa + k0 + 16);
    b0.h[0] = *(const v8h*)(pb0 + k0);
    b0.h[1] = *(const v8h*)(pb0 + k0 + 16);
    b1.h[0] = *(const v8h*)(pb1 + k0);
    b1.h[1] = *(const v8h*)(pb1 + k0 + 16);
    b2.h[0] = *(const v8h*)(pb2 + k0);
    b2.h[1] = *(const v8h*)(pb2 + k0 + 16);
    b3.h[0] = *(const v8h*)(pb3 + k0);
    b3.h[1] = *(const v8h*)(pb3 + k0 + 16);
    // prefetch next K-step (global_prefetch_b8)
    __builtin_prefetch(pa + k0 + 32, 0, 1);
    __builtin_prefetch(pb0 + k0 + 32, 0, 1);
    __builtin_prefetch(pb2 + k0 + 32, 0, 1);
    acc0 = __builtin_amdgcn_wmma_f32_16x16x32_f16(false, af.v, false, b0.v,
                                                  (short)0, acc0, false, false);
    acc1 = __builtin_amdgcn_wmma_f32_16x16x32_f16(false, af.v, false, b1.v,
                                                  (short)0, acc1, false, false);
    acc2 = __builtin_amdgcn_wmma_f32_16x16x32_f16(false, af.v, false, b2.v,
                                                  (short)0, acc2, false, false);
    acc3 = __builtin_amdgcn_wmma_f32_16x16x32_f16(false, af.v, false, b3.v,
                                                  (short)0, acc3, false, false);
  }

  // C layout: vgpr r -> m = row0 + r + 8*hf ; n = col0 + sub*16 + idx
  const int mbase = row0 + (hf ? 8 : 0);
#pragma unroll
  for (int s = 0; s < 4; ++s) {
    const int n = col0 + s * 16 + idx;
    const float badd = bias ? bias[n] : 0.f;
    const v8f a = (s == 0) ? acc0 : (s == 1) ? acc1 : (s == 2) ? acc2 : acc3;
#pragma unroll
    for (int r = 0; r < 8; ++r)
      C[(size_t)(mbase + r) * ldc + n] = a[r] + badd;
  }
}

// ---------------------------------------------------------------------------
// Weight transpose + fp16 convert: W (K,N) row-major -> Wt (N,K) f16.
// ---------------------------------------------------------------------------
__global__ void transpose_h_k(const float* __restrict__ W, _Float16* __restrict__ Wt,
                              int K, int N) {
  int t = blockIdx.x * blockDim.x + threadIdx.x;
  if (t >= K * N) return;
  int k = t / N, n = t - k * N;                  // coalesced read of W
  Wt[(size_t)n * K + k] = (_Float16)W[t];
}

// fp32 -> fp16 convert (layout preserved)
__global__ void tofp16_k(const float* __restrict__ src, _Float16* __restrict__ dst,
                         int n) {
  int t = blockIdx.x * blockDim.x + threadIdx.x;
  if (t < n) dst[t] = (_Float16)src[t];
}

// ---------------------------------------------------------------------------
// im2col (f16 output) for the 3x3 pad-1 conv. x[b,c,y,x] == queries[b,y*7+x,c].
// Acol[row = b*49 + y*7 + x, kk = ic*9 + ky*3 + kx]
// ---------------------------------------------------------------------------
__global__ void im2col_k(const float* __restrict__ q, _Float16* __restrict__ Acol) {
  size_t t = (size_t)blockIdx.x * blockDim.x + threadIdx.x;
  if (t >= (size_t)ROWS * KCONV) return;
  const int kk  = (int)(t % KCONV);
  const int row = (int)(t / KCONV);
  const int b = row / NQ, p = row % NQ;
  const int y = p / 7, x = p % 7;
  const int ic = kk / 9, r = kk % 9;
  const int ky = r / 3, kx = r % 3;
  const int yy = y + ky - 1, xx = x + kx - 1;
  float v = 0.f;
  if (yy >= 0 && yy < 7 && xx >= 0 && xx < 7)
    v = q[((size_t)b * NQ + yy * 7 + xx) * DMODEL + ic];
  Acol[t] = (_Float16)v;
}

// BatchNorm stats over (B, spatial) = 1568 rows per channel (population var).
__global__ void bn_stats_k(const float* __restrict__ h, float* __restrict__ mu,
                           float* __restrict__ rstd) {
  const int c = blockIdx.x, tid = threadIdx.x;
  float s = 0.f, ss = 0.f;
  for (int r = tid; r < ROWS; r += 256) {
    float v = h[(size_t)r * DMODEL + c];
    s += v; ss += v * v;
  }
  __shared__ float sh[256], sh2[256];
  sh[tid] = s; sh2[tid] = ss; __syncthreads();
  for (int off = 128; off; off >>= 1) {
    if (tid < off) { sh[tid] += sh[tid + off]; sh2[tid] += sh2[tid + off]; }
    __syncthreads();
  }
  if (tid == 0) {
    float m = sh[0] * (1.f / ROWS);
    float v = sh2[0] * (1.f / ROWS) - m * m;
    mu[c] = m; rstd[c] = rsqrtf(v + LN_EPS);
  }
}

// xs = relu(BN(h)); eq/ek/ev = f16(xs + queries/keys/values)
__global__ void eqkv_k(const float* __restrict__ h, const float* __restrict__ mu,
                       const float* __restrict__ rstd,
                       const float* __restrict__ q, const float* __restrict__ k,
                       const float* __restrict__ v,
                       _Float16* __restrict__ eq, _Float16* __restrict__ ek,
                       _Float16* __restrict__ ev) {
  size_t t = (size_t)blockIdx.x * blockDim.x + threadIdx.x;
  if (t >= (size_t)ROWS * DMODEL) return;
  const int c = (int)(t % DMODEL);
  float xs = fmaxf((h[t] - mu[c]) * rstd[c], 0.f);
  eq[t] = (_Float16)(xs + q[t]);
  ek[t] = (_Float16)(xs + k[t]);
  ev[t] = (_Float16)(xs + v[t]);
}

// Precompute gate-collapse vectors:
//   u[512] = Wr2 @ Wgate ; wv[14] = rel_embed @ (Wproj @ Wgate)
//   cs = (br2 + bproj) . Wgate + bgate
__global__ void precompute_k(const float* __restrict__ Wr2, const float* __restrict__ Wgate,
                             const float* __restrict__ bgate, const float* __restrict__ Wproj,
                             const float* __restrict__ bproj, const float* __restrict__ br2,
                             const float* __restrict__ rel_embed,
                             float* __restrict__ u, float* __restrict__ wv,
                             float* __restrict__ cs) {
  __shared__ float pv[DKH];
  const int t = threadIdx.x;   // 512 threads
  {
    float s = 0.f;
    for (int j = 0; j < DMODEL; ++j) s += Wr2[(size_t)t * DMODEL + j] * Wgate[j];
    u[t] = s;
  }
  if (t < DKH) {
    float s = 0.f;
    for (int d = 0; d < DMODEL; ++d) s += Wproj[(size_t)t * DMODEL + d] * Wgate[d];
    pv[t] = s;
  }
  __syncthreads();
  if (t < NRELC) {
    float s = 0.f;
    for (int d = 0; d < DKH; ++d) s += rel_embed[t * DKH + d] * pv[d];
    wv[t] = s;
  }
  if (t == 0) {
    float s = bgate[0];
    for (int d = 0; d < DMODEL; ++d) s += (br2[d] + bproj[d]) * Wgate[d];
    cs[0] = s;
  }
}

// att[b,h,qi,ki] = SCALE * q[b,qi,h,:] . k[b,ki,h,:]
__global__ void attqk_k(const float* __restrict__ qh, const float* __restrict__ kh,
                        float* __restrict__ att) {
  size_t t = (size_t)blockIdx.x * blockDim.x + threadIdx.x;
  if (t >= (size_t)BQ * NHEAD * NQ * NQ) return;
  const int ki = (int)(t % NQ);
  const int qi = (int)((t / NQ) % NQ);
  const int h  = (int)((t / (NQ * NQ)) % NHEAD);
  const int b  = (int)(t / ((size_t)NQ * NQ * NHEAD));
  const float* qp = qh + ((size_t)(b * NQ + qi)) * DMODEL + h * DKH;
  const float* kp = kh + ((size_t)(b * NQ + ki)) * DMODEL + h * DKH;
  float s = 0.f;
#pragma unroll 8
  for (int d = 0; d < DKH; ++d) s += qp[d] * kp[d];
  att[t] = s * 0.125f;  // 1/sqrt(64)
}

// One wave32 per (b,qi,ki) pair: LN(256)->ReLU->14 logits->softmax->probs.wv,
// LN(512)->ReLU->dot(u), gate = sigmoid(.)
__global__ void pair_gate_k(const float* __restrict__ A1c, const float* __restrict__ A2c,
                            const float* __restrict__ A1r, const float* __restrict__ A2r,
                            const float* __restrict__ gc, const float* __restrict__ bcln,
                            const float* __restrict__ Wc2, const float* __restrict__ bc2,
                            const float* __restrict__ gr, const float* __restrict__ brln,
                            const float* __restrict__ u, const float* __restrict__ wv,
                            const float* __restrict__ cs, float* __restrict__ gate) {
  const int wave = blockIdx.x * blockDim.y + threadIdx.y;
  if (wave >= NPAIR) return;
  const int lane = threadIdx.x;
  const int b  = wave / (NQ * NQ);
  const int rm = wave % (NQ * NQ);
  const int qi = rm / NQ, ki = rm % NQ;

  // ---- classifier branch: LN over 256, ReLU, 256x14 matvec, softmax ----
  const float* a1 = A1c + (size_t)(b * NQ + qi) * 256;
  const float* a2 = A2c + (size_t)(b * NQ + ki) * 256;
  float x[8], s = 0.f, ss = 0.f;
#pragma unroll
  for (int j = 0; j < 8; ++j) {
    float v = a1[lane + 32 * j] + a2[lane + 32 * j];
    x[j] = v; s += v; ss += v * v;
  }
  for (int off = 16; off; off >>= 1) { s += __shfl_xor(s, off, 32); ss += __shfl_xor(ss, off, 32); }
  float mean = s * (1.f / 256.f);
  float rstd = rsqrtf(ss * (1.f / 256.f) - mean * mean + LN_EPS);

  float lg[NRELC];
#pragma unroll
  for (int r = 0; r < NRELC; ++r) lg[r] = 0.f;
#pragma unroll
  for (int j = 0; j < 8; ++j) {
    const int e = lane + 32 * j;
    float hv = fmaxf((x[j] - mean) * rstd * gc[e] + bcln[e], 0.f);
#pragma unroll
    for (int r = 0; r < NRELC; ++r) lg[r] += hv * Wc2[e * NRELC + r];
  }
  for (int off = 16; off; off >>= 1)
#pragma unroll
    for (int r = 0; r < NRELC; ++r) lg[r] += __shfl_xor(lg[r], off, 32);

  float mx = lg[0] + bc2[0];
#pragma unroll
  for (int r = 0; r < NRELC; ++r) { lg[r] += bc2[r]; mx = fmaxf(mx, lg[r]); }
  float se = 0.f, pw = 0.f;
#pragma unroll
  for (int r = 0; r < NRELC; ++r) {
    float p = __expf(lg[r] - mx);
    se += p; pw += p * wv[r];
  }
  // probs = p/se; top-k(14 of 14) renorm = /(1+1e-8)
  pw /= (se * (1.f + 1e-8f));

  // ---- relation branch: LN over 512, ReLU, dot with u ----
  const float* r1 = A1r + (size_t)(b * NQ + qi) * DMODEL;
  const float* r2 = A2r + (size_t)(b * NQ + ki) * DMODEL;
  float y[16];
  s = 0.f; ss = 0.f;
#pragma unroll
  for (int j = 0; j < 16; ++j) {
    float v = r1[lane + 32 * j] + r2[lane + 32 * j];
    y[j] = v; s += v; ss += v * v;
  }
  for (int off = 16; off; off >>= 1) { s += __shfl_xor(s, off, 32); ss += __shfl_xor(ss, off, 32); }
  mean = s * (1.f / 512.f);
  rstd = rsqrtf(ss * (1.f / 512.f) - mean * mean + LN_EPS);
  float gd = 0.f;
#pragma unroll
  for (int j = 0; j < 16; ++j) {
    const int e = lane + 32 * j;
    float hv = fmaxf((y[j] - mean) * rstd * gr[e] + brln[e], 0.f);
    gd += hv * u[e];
  }
  for (int off = 16; off; off >>= 1) gd += __shfl_xor(gd, off, 32);

  if (lane == 0)
    gate[wave] = 1.f / (1.f + __expf(-(gd + pw + cs[0])));
}

// enh = att*(1+gate); softmax over ki; out[b,qi,h,:] = f16(sum_ki w*v[b,ki,h,:])
__global__ void softmax_av_k(const float* __restrict__ att, const float* __restrict__ gate,
                             const float* __restrict__ vh, _Float16* __restrict__ outp) {
  const int bhq = blockIdx.x;                 // (b*8+h)*49 + qi
  const int b  = bhq / (NHEAD * NQ);
  const int h  = (bhq / NQ) % NHEAD;
  const int qi = bhq % NQ;
  __shared__ float sw[NQ];
  __shared__ float red[2];
  const int tid = threadIdx.x;                // 64 threads
  const float* arow = att + (size_t)bhq * NQ;
  const float* grow = gate + ((size_t)(b * NQ + qi)) * NQ;
  if (tid < NQ) sw[tid] = arow[tid] * (1.f + grow[tid]);
  __syncthreads();
  if (tid == 0) {
    float m = sw[0];
    for (int i = 1; i < NQ; ++i) m = fmaxf(m, sw[i]);
    float sm = 0.f;
    for (int i = 0; i < NQ; ++i) sm += __expf(sw[i] - m);
    red[0] = m; red[1] = sm;
  }
  __syncthreads();
  if (tid < NQ) sw[tid] = __expf(sw[tid] - red[0]) / red[1];
  __syncthreads();
  if (tid < DKH) {
    const float* vbase = vh + (size_t)(b * NQ) * DMODEL + h * DKH + tid;
    float acc = 0.f;
    for (int ki = 0; ki < NQ; ++ki) acc += sw[ki] * vbase[(size_t)ki * DMODEL];
    outp[((size_t)(b * NQ + qi)) * DMODEL + h * DKH + tid] = (_Float16)acc;
  }
}

// ---------------------------------------------------------------------------
static inline void launch_gemm(const _Float16* A, const _Float16* Bt, const float* bias,
                               float* C, int M, int N, int K, int lda, int ldb, int ldc,
                               hipStream_t s) {
  const int tiles = (M / 16) * (N / 64);
  dim3 blk(32, 4), grd((tiles + 3) / 4);
  gemm_wmma_f16<<<grd, blk, 0, s>>>(A, Bt, bias, C, M, N, K, lda, ldb, ldc);
}

extern "C" void kernel_launch(void* const* d_in, const int* in_sizes, int n_in,
                              void* d_out, int out_size, void* d_ws, size_t ws_size,
                              hipStream_t stream) {
  const float* queries = (const float*)d_in[0];
  const float* keys    = (const float*)d_in[1];
  const float* values  = (const float*)d_in[2];
  const float* Wq = (const float*)d_in[3];  const float* bq = (const float*)d_in[4];
  const float* Wk = (const float*)d_in[5];  const float* bk = (const float*)d_in[6];
  const float* Wv = (const float*)d_in[7];  const float* bv = (const float*)d_in[8];
  const float* Wo = (const float*)d_in[9];  const float* bo = (const float*)d_in[10];
  const float* rel_embed = (const float*)d_in[11];
  const float* Wproj = (const float*)d_in[12]; const float* bproj = (const float*)d_in[13];
  const float* Wgate = (const float*)d_in[14]; const float* bgate = (const float*)d_in[15];
  const float* Wc1 = (const float*)d_in[16];   const float* bc1 = (const float*)d_in[17];
  const float* gc = (const float*)d_in[18];    const float* bcln = (const float*)d_in[19];
  const float* Wc2 = (const float*)d_in[20];   const float* bc2 = (const float*)d_in[21];
  const float* Wr1 = (const float*)d_in[22];   const float* br1 = (const float*)d_in[23];
  const float* gr = (const float*)d_in[24];    const float* brln = (const float*)d_in[25];
  const float* Wr2 = (const float*)d_in[26];   const float* br2 = (const float*)d_in[27];
  const float* convW = (const float*)d_in[28]; const float* convb = (const float*)d_in[29];

  // ---- workspace carve-out (256B aligned blocks) ----
  char* base = (char*)d_ws;
  size_t off = 0;
  auto alloc = [&](size_t bytes) -> void* {
    off = (off + 255) & ~(size_t)255;
    void* p = base + off;
    off += bytes;
    return p;
  };
  _Float16* AcolH  = (_Float16*)alloc((size_t)ROWS * KCONV * 2);
  _Float16* convWH = (_Float16*)alloc((size_t)DMODEL * KCONV * 2);
  _Float16* WqT    = (_Float16*)alloc((size_t)DMODEL * DMODEL * 2);
  _Float16* WkT    = (_Float16*)alloc((size_t)DMODEL * DMODEL * 2);
  _Float16* WvT    = (_Float16*)alloc((size_t)DMODEL * DMODEL * 2);
  _Float16* WoT    = (_Float16*)alloc((size_t)DMODEL * DMODEL * 2);
  _Float16* Wc1aT  = (_Float16*)alloc((size_t)256 * DMODEL * 2);
  _Float16* Wc1bT  = (_Float16*)alloc((size_t)256 * DMODEL * 2);
  _Float16* Wr1aT  = (_Float16*)alloc((size_t)DMODEL * DMODEL * 2);
  _Float16* Wr1bT  = (_Float16*)alloc((size_t)DMODEL * DMODEL * 2);
  _Float16* eqH    = (_Float16*)alloc((size_t)ROWS * DMODEL * 2);
  _Float16* ekH    = (_Float16*)alloc((size_t)ROWS * DMODEL * 2);
  _Float16* evH    = (_Float16*)alloc((size_t)ROWS * DMODEL * 2);
  _Float16* attnoH = (_Float16*)alloc((size_t)ROWS * DMODEL * 2);
  float* hconv = (float*)alloc((size_t)ROWS * DMODEL * 4);
  float* muv   = (float*)alloc(DMODEL * 4);
  float* rstdv = (float*)alloc(DMODEL * 4);
  float* qh    = (float*)alloc((size_t)ROWS * DMODEL * 4);
  float* kh    = (float*)alloc((size_t)ROWS * DMODEL * 4);
  float* vh    = (float*)alloc((size_t)ROWS * DMODEL * 4);
  float* A1c   = (float*)alloc((size_t)ROWS * 256 * 4);
  float* A2c   = (float*)alloc((size_t)ROWS * 256 * 4);
  float* A1r   = (float*)alloc((size_t)ROWS * DMODEL * 4);
  float* A2r   = (float*)alloc((size_t)ROWS * DMODEL * 4);
  float* uvec  = (float*)alloc(DMODEL * 4);
  float* wvecp = (float*)alloc(64);
  float* cscal = (float*)alloc(64);
  float* attb  = (float*)alloc((size_t)BQ * NHEAD * NQ * NQ * 4);
  float* gateb = (float*)alloc((size_t)NPAIR * 4);

  // 1) gate-collapse precompute (tiny)
  precompute_k<<<1, 512, 0, stream>>>(Wr2, Wgate, bgate, Wproj, bproj, br2,
                                      rel_embed, uvec, wvecp, cscal);

  // 2) weight transposes / fp16 conversion (one-shot, small)
  {
    int n = DMODEL * DMODEL;
    transpose_h_k<<<(n + 255) / 256, 256, 0, stream>>>(Wq, WqT, DMODEL, DMODEL);
    transpose_h_k<<<(n + 255) / 256, 256, 0, stream>>>(Wk, WkT, DMODEL, DMODEL);
    transpose_h_k<<<(n + 255) / 256, 256, 0, stream>>>(Wv, WvT, DMODEL, DMODEL);
    transpose_h_k<<<(n + 255) / 256, 256, 0, stream>>>(Wo, WoT, DMODEL, DMODEL);
    int n2 = DMODEL * 256;
    transpose_h_k<<<(n2 + 255) / 256, 256, 0, stream>>>(Wc1, Wc1aT, DMODEL, 256);
    transpose_h_k<<<(n2 + 255) / 256, 256, 0, stream>>>(Wc1 + (size_t)DMODEL * 256,
                                                        Wc1bT, DMODEL, 256);
    transpose_h_k<<<(n + 255) / 256, 256, 0, stream>>>(Wr1, Wr1aT, DMODEL, DMODEL);
    transpose_h_k<<<(n + 255) / 256, 256, 0, stream>>>(Wr1 + (size_t)DMODEL * DMODEL,
                                                       Wr1bT, DMODEL, DMODEL);
    int nc = DMODEL * KCONV;
    tofp16_k<<<(nc + 255) / 256, 256, 0, stream>>>(convW, convWH, nc);
  }

  // 3) conv3x3 as im2col + WMMA GEMM (1568 x 512 x 4608)
  {
    size_t tot = (size_t)ROWS * KCONV;
    im2col_k<<<(unsigned)((tot + 255) / 256), 256, 0, stream>>>(queries, AcolH);
  }
  launch_gemm(AcolH, convWH, convb, hconv, ROWS, DMODEL, KCONV, KCONV, KCONV, DMODEL, stream);

  // 4) BatchNorm(affine=False) + ReLU + residual adds (f16 activations)
  bn_stats_k<<<DMODEL, 256, 0, stream>>>(hconv, muv, rstdv);
  eqkv_k<<<(ROWS * DMODEL) / 256, 256, 0, stream>>>(hconv, muv, rstdv, queries, keys,
                                                    values, eqH, ekH, evH);

  // 5) token-level projections (all WMMA)
  launch_gemm(eqH, WqT, bq, qh, ROWS, DMODEL, DMODEL, DMODEL, DMODEL, DMODEL, stream);
  launch_gemm(ekH, WkT, bk, kh, ROWS, DMODEL, DMODEL, DMODEL, DMODEL, DMODEL, stream);
  launch_gemm(evH, WvT, bv, vh, ROWS, DMODEL, DMODEL, DMODEL, DMODEL, DMODEL, stream);
  launch_gemm(eqH, Wc1aT, bc1,     A1c, ROWS, 256,    DMODEL, DMODEL, DMODEL, 256,    stream);
  launch_gemm(ekH, Wc1bT, nullptr, A2c, ROWS, 256,    DMODEL, DMODEL, DMODEL, 256,    stream);
  launch_gemm(eqH, Wr1aT, br1,     A1r, ROWS, DMODEL, DMODEL, DMODEL, DMODEL, DMODEL, stream);
  launch_gemm(ekH, Wr1bT, nullptr, A2r, ROWS, DMODEL, DMODEL, DMODEL, DMODEL, DMODEL, stream);

  // 6) attention logits
  {
    size_t tot = (size_t)BQ * NHEAD * NQ * NQ;
    attqk_k<<<(unsigned)((tot + 255) / 256), 256, 0, stream>>>(qh, kh, attb);
  }

  // 7) per-pair gate (one wave32 per pair)
  {
    dim3 blk(32, 8), grd((NPAIR + 7) / 8);
    pair_gate_k<<<grd, blk, 0, stream>>>(A1c, A2c, A1r, A2r, gc, bcln, Wc2, bc2,
                                         gr, brln, uvec, wvecp, cscal, gateb);
  }

  // 8) gated softmax + A*V (f16 output feeds final GEMM)
  softmax_av_k<<<BQ * NHEAD * NQ, 64, 0, stream>>>(attb, gateb, vh, attnoH);

  // 9) output projection (WMMA) straight into d_out
  launch_gemm(attnoH, WoT, bo, (float*)d_out, ROWS, DMODEL, DMODEL, DMODEL, DMODEL,
              DMODEL, stream);
}
